// SPGAT_80719615361789
// MI455X (gfx1250) — matrix operand order repaired and checked by reading, not compiled, not run on hardware
//
#include <hip/hip_runtime.h>

typedef __bf16 bh16;
typedef __attribute__((ext_vector_type(8)))  __bf16 v8bf;
typedef __attribute__((ext_vector_type(16))) __bf16 v16bf;
typedef __attribute__((ext_vector_type(8)))  float  v8f;

constexpr int   NN  = 50000;   // nodes
constexpr int   EE  = 800000;  // edges
constexpr int   FIN = 256;     // input features
constexpr int   HID = 64;      // hidden per head
constexpr int   NHD = 8;       // heads layer 1
constexpr int   NC  = 40;      // classes
constexpr int   NCP = 48;      // classes padded to 3x16 tiles
constexpr int   F2  = NHD * HID; // 512, layer-2 input width
constexpr int   BPAD = 8;      // LDS row pad (bf16 elems) -> 4-word bank rotation
constexpr float LRA = 0.2f;    // leaky relu slope

// ---------- ordered-uint encoding for float atomic max ----------
__device__ __forceinline__ unsigned f2ord(float f) {
  unsigned u = __float_as_uint(f);
  return (u & 0x80000000u) ? ~u : (u | 0x80000000u);
}
__device__ __forceinline__ float ord2f(unsigned u) {
  unsigned b = (u & 0x80000000u) ? (u & 0x7fffffffu) : ~u;
  return __uint_as_float(b);
}

// ---------- casts / transposes ----------
__global__ void k_f2bf(const float* __restrict__ in, bh16* __restrict__ out, int n) {
  int i = blockIdx.x * blockDim.x + threadIdx.x;
  if (i < n) out[i] = (bh16)in[i];
}

// W1 [8,256,64] -> W1t [8,64,256] bf16
__global__ void k_castw1(const float* __restrict__ W1, bh16* __restrict__ Wt) {
  int i = blockIdx.x * blockDim.x + threadIdx.x;
  if (i >= NHD * HID * FIN) return;
  int h = i / (HID * FIN);
  int r = i % (HID * FIN);
  int o = r / FIN;
  int k = r % FIN;
  Wt[i] = (bh16)W1[(h * FIN + k) * HID + o];
}

// W2 [512,40] -> W2t [48,512] bf16, zero padded
__global__ void k_castw2(const float* __restrict__ W2, bh16* __restrict__ Wt) {
  int i = blockIdx.x * blockDim.x + threadIdx.x;
  if (i >= NCP * F2) return;
  int c = i / F2;
  int k = i % F2;
  Wt[i] = (bh16)((c < NC) ? W2[k * NC + c] : 0.f);
}

__global__ void k_initout(float* __restrict__ out, const float* __restrict__ b2) {
  int i = blockIdx.x * blockDim.x + threadIdx.x;
  if (i < NN * NC) out[i] = b2[i % NC];
}

// ---------- fragment builders ----------
__device__ __forceinline__ v16bf join16(v8bf lo, v8bf hg) {
  v16bf a;
#pragma unroll
  for (int i = 0; i < 8; i++) { a[i] = lo[i]; a[i + 8] = hg[i]; }
  return a;
}

// ---------- layer-1 GEMM: H1[h] = Xbf @ W1[h] ----------
// Block = 8 waves, one head (blockIdx.y). B for the head (64x256 bf16 = 32KB)
// is staged once into LDS (padded rows for bank rotation); each wave computes a
// 16x64 tile: B fragments via ds_load, A double-buffered from global.
__global__ __launch_bounds__(256) void k_gemm1(const bh16* __restrict__ X,
                                               const bh16* __restrict__ Wt,
                                               bh16* __restrict__ Hout) {
  __shared__ bh16 Bs[HID][FIN + BPAD];
  int head = blockIdx.y;
  const bh16* Bh = Wt + (size_t)head * HID * FIN;
  // cooperative global -> LDS stage of the whole head B matrix
  for (int idx = threadIdx.x; idx < HID * (FIN / 8); idx += 256) {
    int o  = idx >> 5;          // row 0..63   (FIN/8 == 32 chunks per row)
    int k8 = (idx & 31) * 8;    // 16B chunk within the row
    *(v8bf*)&Bs[o][k8] = *(const v8bf*)(Bh + (size_t)o * FIN + k8);
  }
  __syncthreads();

  int wv   = threadIdx.x >> 5;
  int lane = threadIdx.x & 31;
  int mt = blockIdx.x * 8 + wv;
  if (mt >= NN / 16) return;    // after the only barrier: safe
  int m0 = mt * 16;
  int hi = lane >> 4;           // lane half selects K sub-blocks per ISA layout
  int lc = lane & 15;
  const bh16* Arow = X + (size_t)(m0 + lc) * FIN;

  v8f acc[4];
#pragma unroll
  for (int t = 0; t < 4; t++)
#pragma unroll
    for (int i = 0; i < 8; i++) acc[t][i] = 0.f;

  // A fragment double buffer (global)
  v8bf aLo = *(const v8bf*)(Arow + hi * 8);
  v8bf aHi = *(const v8bf*)(Arow + hi * 8 + 16);

#pragma unroll
  for (int kc = 0; kc < FIN; kc += 32) {
    v16bf a = join16(aLo, aHi);
    if (kc + 32 < FIN) {        // issue next chunk's A loads before the WMMAs
      aLo = *(const v8bf*)(Arow + kc + 32 + hi * 8);
      aHi = *(const v8bf*)(Arow + kc + 32 + hi * 8 + 16);
    }
#pragma unroll
    for (int t = 0; t < 4; t++) {
      int col = t * 16 + lc;    // B fragment: col fixed, K = 8hi..+7 and 16+8hi..+7 (LDS)
      v16bf b = join16(*(const v8bf*)&Bs[col][kc + hi * 8],
                       *(const v8bf*)&Bs[col][kc + hi * 8 + 16]);
      acc[t] = __builtin_amdgcn_wmma_f32_16x16x32_bf16(
          false, a, false, b, (short)0, acc[t], false, false);
    }
  }
  // C/D layout: vgpr r -> row m0 + r + 8*hi, col = t*16 + lc
#pragma unroll
  for (int t = 0; t < 4; t++)
#pragma unroll
    for (int r = 0; r < 8; r++)
      Hout[((size_t)head * NN + m0 + r + 8 * hi) * HID + t * 16 + lc] =
          (bh16)acc[t][r];
}

// ---------- layer-2 GEMM: H2 = h2bf @ W2t  (B 48x512 bf16 staged in LDS) ----------
__global__ __launch_bounds__(256) void k_gemm2(const bh16* __restrict__ X,
                                               const bh16* __restrict__ Wt,
                                               float* __restrict__ Hout) {
  __shared__ bh16 Bs[NCP][F2 + BPAD];
  for (int idx = threadIdx.x; idx < NCP * (F2 / 8); idx += 256) {
    int o  = idx >> 6;          // row 0..47   (F2/8 == 64 chunks per row)
    int k8 = (idx & 63) * 8;
    *(v8bf*)&Bs[o][k8] = *(const v8bf*)(Wt + (size_t)o * F2 + k8);
  }
  __syncthreads();

  int wv   = threadIdx.x >> 5;
  int lane = threadIdx.x & 31;
  int mt = blockIdx.x * 8 + wv;
  if (mt >= NN / 16) return;
  int m0 = mt * 16;
  int hi = lane >> 4;
  int lc = lane & 15;
  const bh16* Arow = X + (size_t)(m0 + lc) * F2;

  v8f acc[3];
#pragma unroll
  for (int t = 0; t < 3; t++)
#pragma unroll
    for (int i = 0; i < 8; i++) acc[t][i] = 0.f;

  v8bf aLo = *(const v8bf*)(Arow + hi * 8);
  v8bf aHi = *(const v8bf*)(Arow + hi * 8 + 16);

#pragma unroll
  for (int kc = 0; kc < F2; kc += 32) {
    v16bf a = join16(aLo, aHi);
    if (kc + 32 < F2) {
      aLo = *(const v8bf*)(Arow + kc + 32 + hi * 8);
      aHi = *(const v8bf*)(Arow + kc + 32 + hi * 8 + 16);
    }
#pragma unroll
    for (int t = 0; t < 3; t++) {
      int col = t * 16 + lc;
      v16bf b = join16(*(const v8bf*)&Bs[col][kc + hi * 8],
                       *(const v8bf*)&Bs[col][kc + hi * 8 + 16]);
      acc[t] = __builtin_amdgcn_wmma_f32_16x16x32_bf16(
          false, a, false, b, (short)0, acc[t], false, false);
    }
  }
#pragma unroll
  for (int t = 0; t < 3; t++)
#pragma unroll
    for (int r = 0; r < 8; r++)
      Hout[(size_t)(m0 + r + 8 * hi) * NCP + t * 16 + lc] = acc[t][r];
}

// ---------- per-node attention scalars ----------
__global__ void k_rowdot1(const bh16* __restrict__ H1, const float* __restrict__ a1,
                          float* __restrict__ ssrc, float* __restrict__ stgt) {
  int wid = (blockIdx.x * blockDim.x + threadIdx.x) >> 5;
  int lane = threadIdx.x & 31;
  if (wid >= NN * NHD) return;
  int h = wid & (NHD - 1);
  int n = wid >> 3;
  const bh16* row = H1 + ((size_t)h * NN + n) * HID;
  float p = 0.f, q = 0.f;
#pragma unroll
  for (int o = 0; o < HID; o += 32) {
    float v = (float)row[o + lane];
    p += v * a1[h * 2 * HID + o + lane];
    q += v * a1[h * 2 * HID + HID + o + lane];
  }
  for (int off = 16; off; off >>= 1) { p += __shfl_xor(p, off); q += __shfl_xor(q, off); }
  if (lane == 0) { ssrc[n * NHD + h] = p; stgt[n * NHD + h] = q; }
}

__global__ void k_rowdot2(const float* __restrict__ H2, const float* __restrict__ a2,
                          float* __restrict__ ssrc, float* __restrict__ stgt) {
  int wid = (blockIdx.x * blockDim.x + threadIdx.x) >> 5;
  int lane = threadIdx.x & 31;
  if (wid >= NN) return;
  const float* row = H2 + (size_t)wid * NCP;
  float p = row[lane] * a2[lane];
  float q = row[lane] * a2[NC + lane];
  if (lane < NC - 32) {
    p += row[32 + lane] * a2[32 + lane];
    q += row[32 + lane] * a2[NC + 32 + lane];
  }
  for (int off = 16; off; off >>= 1) { p += __shfl_xor(p, off); q += __shfl_xor(q, off); }
  if (lane == 0) { ssrc[wid] = p; stgt[wid] = q; }
}

// ---------- edge softmax: global max + denom ----------
template <int H>
__global__ void k_edge_max(const int* __restrict__ src, const int* __restrict__ tgt,
                           const float* __restrict__ ssrc, const float* __restrict__ stgt,
                           unsigned* __restrict__ maxbuf) {
  __shared__ unsigned sm[H];
  if (threadIdx.x < H) sm[threadIdx.x] = 0u;
  __syncthreads();
  int e = blockIdx.x * blockDim.x + threadIdx.x;
  int s = 0, t = 0;
  bool ok = e < EE;
  if (ok) { s = src[e]; t = tgt[e]; }
#pragma unroll
  for (int h = 0; h < H; h++) {
    float v = ok ? (ssrc[s * H + h] + stgt[t * H + h]) : -__builtin_inff();
    v = v > 0.f ? v : LRA * v;
    for (int o = 16; o; o >>= 1) v = fmaxf(v, __shfl_xor(v, o));
    if ((threadIdx.x & 31) == 0) atomicMax(&sm[h], f2ord(v));
  }
  __syncthreads();
  if (threadIdx.x < H) atomicMax(&maxbuf[threadIdx.x], sm[threadIdx.x]);
}

template <int H>
__global__ void k_edge_denom(const int* __restrict__ src, const int* __restrict__ tgt,
                             const float* __restrict__ ssrc, const float* __restrict__ stgt,
                             const unsigned* __restrict__ maxbuf,
                             float* __restrict__ denom) {
  int e = blockIdx.x * blockDim.x + threadIdx.x;
  if (e >= EE) return;
  int s = src[e], t = tgt[e];
#pragma unroll
  for (int h = 0; h < H; h++) {
    float v = ssrc[s * H + h] + stgt[t * H + h];
    v = v > 0.f ? v : LRA * v;
    unsafeAtomicAdd(&denom[s * H + h], __expf(v - ord2f(maxbuf[h])));
  }
}

// ---------- layer-1 weighted scatter-add (wave per edge, 8 heads x 64) ----------
__global__ void k_agg1(const int* __restrict__ src, const int* __restrict__ tgt,
                       const float* __restrict__ ssrc, const float* __restrict__ stgt,
                       const unsigned* __restrict__ maxbuf, const float* __restrict__ denom,
                       const bh16* __restrict__ H1, float* __restrict__ agg) {
  int wid = (blockIdx.x * blockDim.x + threadIdx.x) >> 5;
  int lane = threadIdx.x & 31;
  if (wid >= EE) return;
  int s = src[wid], t = tgt[wid];
  // warm the gather stream for head 0
  __builtin_prefetch(H1 + (size_t)t * HID + lane * 2, 0, 3);
#pragma unroll
  for (int h = 0; h < NHD; h++) {
    if (h + 1 < NHD)  // prefetch next head's row while this head computes
      __builtin_prefetch(H1 + ((size_t)(h + 1) * NN + t) * HID + lane * 2, 0, 3);
    float v = ssrc[s * NHD + h] + stgt[t * NHD + h];
    v = v > 0.f ? v : LRA * v;
    float att = __expf(v - ord2f(maxbuf[h])) / (denom[s * NHD + h] + 1e-10f);
    const bh16* hv = H1 + ((size_t)h * NN + t) * HID + lane * 2;
    float* op = agg + (size_t)s * F2 + h * HID + lane * 2;
    unsafeAtomicAdd(op,     att * (float)hv[0]);
    unsafeAtomicAdd(op + 1, att * (float)hv[1]);
  }
}

// ---------- layer-2 weighted scatter-add ----------
__global__ void k_agg2(const int* __restrict__ src, const int* __restrict__ tgt,
                       const float* __restrict__ ssrc, const float* __restrict__ stgt,
                       const unsigned* __restrict__ maxbuf, const float* __restrict__ denom,
                       const float* __restrict__ H2, float* __restrict__ out) {
  int wid = (blockIdx.x * blockDim.x + threadIdx.x) >> 5;
  int lane = threadIdx.x & 31;
  if (wid >= EE) return;
  int s = src[wid], t = tgt[wid];
  float v = ssrc[s] + stgt[t];
  v = v > 0.f ? v : LRA * v;
  float att = __expf(v - ord2f(maxbuf[0])) / (denom[s] + 1e-10f);
  for (int c = lane; c < NC; c += 32)
    unsafeAtomicAdd(&out[(size_t)s * NC + c], att * H2[(size_t)t * NCP + c]);
}

// ---------- ELU + concat-cast for layer-2 input ----------
__global__ void k_elu(const float* __restrict__ agg, const float* __restrict__ b1,
                      bh16* __restrict__ h2bf) {
  int i = blockIdx.x * blockDim.x + threadIdx.x;
  if (i >= NN * F2) return;
  float v = agg[i] + b1[i & (F2 - 1)];
  v = v > 0.f ? v : (__expf(v) - 1.f);
  h2bf[i] = (bh16)v;
}

// ---------- final log-softmax (warp per node, in-place) ----------
__global__ void k_lsm(float* __restrict__ out) {
  int wid = (blockIdx.x * blockDim.x + threadIdx.x) >> 5;
  int lane = threadIdx.x & 31;
  if (wid >= NN) return;
  float* row = out + (size_t)wid * NC;
  float v0 = row[lane];
  float v1 = (lane < NC - 32) ? row[32 + lane] : -__builtin_inff();
  float m = fmaxf(v0, v1);
  for (int o = 16; o; o >>= 1) m = fmaxf(m, __shfl_xor(m, o));
  float sum = __expf(v0 - m) + ((lane < NC - 32) ? __expf(v1 - m) : 0.f);
  for (int o = 16; o; o >>= 1) sum += __shfl_xor(sum, o);
  float lse = m + __logf(sum);
  row[lane] = v0 - lse;
  if (lane < NC - 32) row[32 + lane] = v1 - lse;
}

// ---------- workspace layout (bytes) ----------
constexpr size_t OFF_XBF  = 0;                               // 25.6 MB  bf16 x
constexpr size_t OFF_W1T  = OFF_XBF  + (size_t)NN * FIN * 2; // 256 KB
constexpr size_t OFF_W2T  = OFF_W1T  + (size_t)NHD * HID * FIN * 2;
constexpr size_t OFF_H1   = OFF_W2T  + (size_t)NCP * F2 * 2; // 51.2 MB bf16 H1
constexpr size_t OFF_S1S  = OFF_H1   + (size_t)NHD * NN * HID * 2;
constexpr size_t OFF_S1T  = OFF_S1S  + (size_t)NN * NHD * 4;
constexpr size_t OFF_MAX1 = OFF_S1T  + (size_t)NN * NHD * 4;
constexpr size_t OFF_DEN1 = OFF_MAX1 + 256;
constexpr size_t OFF_AGG1 = OFF_DEN1 + (size_t)NN * NHD * 4; // 102.4 MB f32
constexpr size_t OFF_H2BF = OFF_AGG1 + (size_t)NN * F2 * 4;  // 51.2 MB bf16
constexpr size_t OFF_H2F  = OFF_H2BF + (size_t)NN * F2 * 2;  // 9.6 MB f32
constexpr size_t OFF_S2S  = OFF_H2F  + (size_t)NN * NCP * 4;
constexpr size_t OFF_S2T  = OFF_S2S  + (size_t)NN * 4;
constexpr size_t OFF_MAX2 = OFF_S2T  + (size_t)NN * 4;
constexpr size_t OFF_DEN2 = OFF_MAX2 + 256;

extern "C" void kernel_launch(void* const* d_in, const int* in_sizes, int n_in,
                              void* d_out, int out_size, void* d_ws, size_t ws_size,
                              hipStream_t stream) {
  const float* x  = (const float*)d_in[0];
  const int*   el = (const int*)  d_in[1];
  const float* W1 = (const float*)d_in[2];
  const float* a1 = (const float*)d_in[3];
  const float* b1 = (const float*)d_in[4];
  const float* W2 = (const float*)d_in[5];
  const float* a2 = (const float*)d_in[6];
  const float* b2 = (const float*)d_in[7];
  const int* src = el;
  const int* tgt = el + EE;
  float* out = (float*)d_out;
  char*  ws  = (char*)d_ws;

  bh16*     Xbf  = (bh16*)    (ws + OFF_XBF);
  bh16*     W1t  = (bh16*)    (ws + OFF_W1T);
  bh16*     W2t  = (bh16*)    (ws + OFF_W2T);
  bh16*     H1   = (bh16*)    (ws + OFF_H1);
  float*    s1s  = (float*)   (ws + OFF_S1S);
  float*    s1t  = (float*)   (ws + OFF_S1T);
  unsigned* max1 = (unsigned*)(ws + OFF_MAX1);
  float*    den1 = (float*)   (ws + OFF_DEN1);
  float*    agg1 = (float*)   (ws + OFF_AGG1);
  bh16*     h2bf = (bh16*)    (ws + OFF_H2BF);
  float*    H2f  = (float*)   (ws + OFF_H2F);
  float*    s2s  = (float*)   (ws + OFF_S2S);
  float*    s2t  = (float*)   (ws + OFF_S2T);
  unsigned* max2 = (unsigned*)(ws + OFF_MAX2);
  float*    den2 = (float*)   (ws + OFF_DEN2);

  // zero accumulators (capture-safe memsets)
  hipMemsetAsync(max1, 0, 256, stream);
  hipMemsetAsync(max2, 0, 256, stream);
  hipMemsetAsync(den1, 0, (size_t)NN * NHD * 4, stream);
  hipMemsetAsync(den2, 0, (size_t)NN * 4, stream);
  hipMemsetAsync(agg1, 0, (size_t)NN * F2 * 4, stream);

  // casts / transposes / output bias init
  k_f2bf<<<(NN * FIN + 255) / 256, 256, 0, stream>>>(x, Xbf, NN * FIN);
  k_castw1<<<(NHD * HID * FIN + 255) / 256, 256, 0, stream>>>(W1, W1t);
  k_castw2<<<(NCP * F2 + 255) / 256, 256, 0, stream>>>(W2, W2t);
  k_initout<<<(NN * NC + 255) / 256, 256, 0, stream>>>(out, b2);

  // layer 1
  k_gemm1<<<dim3((NN / 16 + 7) / 8, NHD), 256, 0, stream>>>(Xbf, W1t, H1);
  k_rowdot1<<<(NN * NHD * 32) / 256, 256, 0, stream>>>(H1, a1, s1s, s1t);
  k_edge_max<NHD><<<EE / 256, 256, 0, stream>>>(src, tgt, s1s, s1t, max1);
  k_edge_denom<NHD><<<EE / 256, 256, 0, stream>>>(src, tgt, s1s, s1t, max1, den1);
  k_agg1<<<(EE * 32) / 256, 256, 0, stream>>>(src, tgt, s1s, s1t, max1, den1, H1, agg1);
  k_elu<<<(NN * F2 + 255) / 256, 256, 0, stream>>>(agg1, b1, h2bf);

  // layer 2
  k_gemm2<<<(NN / 16 + 7) / 8, 256, 0, stream>>>(h2bf, W2t, H2f);
  k_rowdot2<<<(NN * 32) / 256, 256, 0, stream>>>(H2f, a2, s2s, s2t);
  k_edge_max<1><<<EE / 256, 256, 0, stream>>>(src, tgt, s2s, s2t, max2);
  k_edge_denom<1><<<EE / 256, 256, 0, stream>>>(src, tgt, s2s, s2t, max2, den2);
  k_agg2<<<(EE * 32) / 256, 256, 0, stream>>>(src, tgt, s2s, s2t, max2, den2, H2f, out);
  k_lsm<<<(NN * 32) / 256, 256, 0, stream>>>(out);
}
